// Attention_11836929868370
// MI455X (gfx1250) — compile-verified, matrix operand order checked by reading, not run on hardware
//
#include <hip/hip_runtime.h>
#include <hip/hip_bf16.h>

typedef __attribute__((ext_vector_type(16))) __bf16       v16bf;
typedef __attribute__((ext_vector_type(8)))  float        v8f;
typedef __attribute__((ext_vector_type(8)))  unsigned int v8u;

#define DIM   512
#define NH    8
#define HD    64
#define SEQ   2048
#define BATCH 4
#define QKVD  1536

__device__ __forceinline__ unsigned short f2bfbits(float f) {
  union { float f; unsigned u; } a; a.f = f;
  unsigned u = a.u;
  return (unsigned short)((u + 0x7FFFu + ((u >> 16) & 1u)) >> 16);   // RNE
}
__device__ __forceinline__ unsigned pack2bf(float lo, float hi) {
  return (unsigned)f2bfbits(lo) | ((unsigned)f2bfbits(hi) << 16);
}
// K index of bf16 pair p (0..7) for this half-wave, per CDNA5 16-bit A/B layout
__device__ __forceinline__ int kOfPair(int p, int half) {
  return ((p < 4) ? (2 * p) : (16 + 2 * (p - 4))) + 8 * half;
}

// ---------------------------------------------------------------------------
// Kernel 0: bulk f32 -> bf16 conversion (done ONCE; GEMM loops stay pure
// load+wmma instead of re-packing operands per tile)
// ---------------------------------------------------------------------------
__global__ __launch_bounds__(256) void cvt_bf16_kernel(
    const float* __restrict__ src, unsigned* __restrict__ dst, int npairs)
{
  int i = blockIdx.x * 256 + threadIdx.x;
  const int stride = gridDim.x * 256;
  for (; i < npairs; i += stride) {
    float2 v = ((const float2*)src)[i];
    dst[i] = pack2bf(v.x, v.y);
  }
}

// ---------------------------------------------------------------------------
// Kernel 1: qkv = x @ Wqkv^T + b; per-head LayerNorm + RoPE on q/k; q *= HD^-0.5
//           q,k -> bf16 [B,NH,N,HD];  v -> bf16 transposed [B,NH,HD,N]
//           A (x) and B (Wqkv) already bf16 in workspace.
// ---------------------------------------------------------------------------
__global__ __launch_bounds__(256) void qkv_ln_rope_kernel(
    const unsigned short* __restrict__ xbf, const unsigned short* __restrict__ Wbf,
    const float* __restrict__ bias,
    const float* __restrict__ qn_g, const float* __restrict__ qn_b,
    const float* __restrict__ kn_g, const float* __restrict__ kn_b,
    unsigned short* __restrict__ qws, unsigned short* __restrict__ kws,
    unsigned short* __restrict__ vTws)
{
  __shared__ float tile[32 * 65];
  const int tid  = threadIdx.x;
  const int wave = tid >> 5, lane = tid & 31;
  const int wm = wave >> 2, wn = wave & 3;
  const int half = lane >> 4, nl = lane & 15;
  const int row0 = blockIdx.y * 32;
  const int col0 = blockIdx.x * 64;

  const int rA = row0 + wm * 16 + nl;   // A-fragment row (lane = row)
  const int cB = col0 + wn * 16 + nl;   // B-fragment column (lane = col)
  const unsigned* xrow = (const unsigned*)(xbf + (size_t)rA * DIM);
  const unsigned* wrow = (const unsigned*)(Wbf + (size_t)cB * DIM);

  v8f acc = {};
  for (int kb = 0; kb < DIM; kb += 32) {
    v8u au, bu;
#pragma unroll
    for (int p = 0; p < 8; ++p) {
      int k = kb + kOfPair(p, half);
      au[p] = xrow[k >> 1];
      bu[p] = wrow[k >> 1];
    }
    acc = __builtin_amdgcn_wmma_f32_16x16x32_bf16(
        false, __builtin_bit_cast(v16bf, au),
        false, __builtin_bit_cast(v16bf, bu),
        (short)0, acc, false, false);
  }
  const float bc = bias[cB];
#pragma unroll
  for (int r = 0; r < 8; ++r) {             // D layout: lane col = nl, rows half*8+r
    int ml = wm * 16 + half * 8 + r;
    tile[ml * 65 + (wn * 16 + nl)] = acc[r] + bc;
  }
  __syncthreads();

  const int which = col0 / DIM;             // 0=q 1=k 2=v (64-col tile = one head)
  const int h = (col0 % DIM) / HD;
  if (tid < 32) {
    const int gm = row0 + tid;
    const int b = gm >> 11;                 // / SEQ
    const int n = gm & (SEQ - 1);
    float v[64];
#pragma unroll
    for (int d = 0; d < 64; ++d) v[d] = tile[tid * 65 + d];

    if (which == 2) {                       // V: store transposed [.,HD,N]
      unsigned short* dst = vTws + (size_t)(b * NH + h) * HD * SEQ + n;
#pragma unroll
      for (int d = 0; d < 64; ++d) dst[(size_t)d * SEQ] = f2bfbits(v[d]);
    } else {
      const float* g  = (which == 0) ? qn_g : kn_g;
      const float* bb = (which == 0) ? qn_b : kn_b;
      float mu = 0.f;
#pragma unroll
      for (int d = 0; d < 64; ++d) mu += v[d];
      mu *= (1.f / 64.f);
      float var = 0.f;
#pragma unroll
      for (int d = 0; d < 64; ++d) { float t = v[d] - mu; var += t * t; }
      var *= (1.f / 64.f);
      const float istd = rsqrtf(var + 1e-6f);
#pragma unroll
      for (int d = 0; d < 64; ++d) v[d] = (v[d] - mu) * istd * g[d] + bb[d];
      const float scale = (which == 0) ? 0.125f : 1.f;   // HD^-0.5 folded into q
      unsigned* dst = (unsigned*)(((which == 0) ? qws : kws) +
                                  ((size_t)(b * NH + h) * SEQ + n) * HD);
#pragma unroll
      for (int i = 0; i < 32; ++i) {        // RoPE pair (2i, 2i+1)
        float ang = (float)n * __expf(-(float)i * (9.210340372f / 32.f));
        float s, c;
        __sincosf(ang, &s, &c);
        float y0 = v[2 * i], y1 = v[2 * i + 1];
        dst[i] = pack2bf((y0 * c - y1 * s) * scale, (y1 * c + y0 * s) * scale);
      }
    }
  }
}

// ---------------------------------------------------------------------------
// Kernel 2: per (b,h, 32-row q-tile): S = q.k^T (masked) -> exact softmax over
// full 2048-wide row in 256KB LDS. During normalization each thread packs its
// own 256-col chunk to bf16 *in place* (chunk-local layout, race-free), so the
// P.V pass reads ready-made b32 bf16 pairs from LDS. Output bf16 [B,N,DIM].
// ---------------------------------------------------------------------------
__global__ __launch_bounds__(256) void attn_kernel(
    const unsigned short* __restrict__ qws, const unsigned short* __restrict__ kws,
    const unsigned short* __restrict__ vTws, const unsigned char* __restrict__ mask,
    unsigned short* __restrict__ obf)
{
  extern __shared__ float S[];              // 32 x SEQ scores
  __shared__ float red[32 * 8];
  const int tid  = threadIdx.x;
  const int wave = tid >> 5, lane = tid & 31;
  const int wm = wave >> 2, wn = wave & 3;
  const int half = lane >> 4, nl = lane & 15;
  const int bh = blockIdx.y;
  const int b = bh >> 3, h = bh & 7;
  const int row0 = blockIdx.x * 32;

  const unsigned* qbase = (const unsigned*)(qws + (size_t)bh * SEQ * HD);
  const unsigned* kbase = (const unsigned*)(kws + (size_t)bh * SEQ * HD);
  const unsigned* vbase = (const unsigned*)(vTws + (size_t)bh * HD * SEQ);
  const unsigned char* mrow = mask + (size_t)b * SEQ;

  // ---- Pass 1: scores ----
  const int qrow = row0 + wm * 16 + nl;
  for (int kvb = 0; kvb < SEQ; kvb += 64) {
    const int j0 = kvb + wn * 16;
    if (kvb + 64 < SEQ)                     // stream next K block into cache
      __builtin_prefetch(kbase + ((j0 + 64 + nl) * HD >> 1), 0, 1);
    v8f acc = {};
#pragma unroll
    for (int dstep = 0; dstep < HD; dstep += 32) {
      v8u au, bu;
#pragma unroll
      for (int p = 0; p < 8; ++p) {
        int k = dstep + kOfPair(p, half);
        au[p] = qbase[(qrow * HD + k) >> 1];
        bu[p] = kbase[((j0 + nl) * HD + k) >> 1];
      }
      acc = __builtin_amdgcn_wmma_f32_16x16x32_bf16(
          false, __builtin_bit_cast(v16bf, au),
          false, __builtin_bit_cast(v16bf, bu),
          (short)0, acc, false, false);
    }
    const int j = j0 + nl;
    const bool ok = mrow[j] != 0;
#pragma unroll
    for (int r = 0; r < 8; ++r) {
      int ml = wm * 16 + half * 8 + r;
      S[ml * SEQ + j] = ok ? acc[r] : -__builtin_inff();
    }
  }
  __syncthreads();

  // ---- Pass 2: exact softmax, 8 threads per row; pack P->bf16 in place ----
  {
    const int row = tid >> 3, sub = tid & 7;
    float* srow = S + row * SEQ;
    const int c0 = sub * 256;               // this thread's own chunk
    float lmax = -__builtin_inff();
    for (int c = 0; c < 256; ++c) lmax = fmaxf(lmax, srow[c0 + c]);
    red[row * 8 + sub] = lmax;
    __syncthreads();
    float rm = -__builtin_inff();
#pragma unroll
    for (int i = 0; i < 8; ++i) rm = fmaxf(rm, red[row * 8 + i]);
    float ls = 0.f;
    for (int c = 0; c < 256; ++c) {
      float e = __expf(srow[c0 + c] - rm);
      srow[c0 + c] = e;
      ls += e;
    }
    __syncthreads();
    red[row * 8 + sub] = ls;
    __syncthreads();
    float rs = 0.f;
#pragma unroll
    for (int i = 0; i < 8; ++i) rs += red[row * 8 + i];
    const float inv = 1.f / rs;
    // Pack normalized probs for columns [c0, c0+256) into the first 128 float
    // slots of this same chunk (uint index c0 + c/2). Chunk-local => no
    // cross-thread overlap; ascending order => write idx <= read idx.
    unsigned* prow = (unsigned*)srow;
    for (int c = 0; c < 256; c += 2) {
      float e0 = srow[c0 + c], e1 = srow[c0 + c + 1];
      asm volatile("" ::: "memory");        // pin read-before-write ordering
      prow[c0 + (c >> 1)] = pack2bf(e0 * inv, e1 * inv);
    }
  }
  __syncthreads();

  // ---- Pass 3: O = P.V (P packed bf16 in LDS, V pre-transposed in global) ----
  {
    v8f acc = {};
    const int arow = wm * 16 + nl;
    const int dcol = wn * 16 + nl;
    const unsigned* prow = (const unsigned*)S + (size_t)arow * SEQ;
    for (int jb = 0; jb < SEQ; jb += 32) {
      const int chunkbase = (jb >> 8) << 8;   // uint base of 256-col chunk
      const int jo = jb & 255;
      if (jb + 32 < SEQ)
        __builtin_prefetch(vbase + ((dcol * SEQ + jb + 32) >> 1), 0, 1);
      v8u au, bu;
#pragma unroll
      for (int p = 0; p < 8; ++p) {
        int kk = kOfPair(p, half);
        au[p] = prow[chunkbase + ((jo + kk) >> 1)];
        bu[p] = vbase[(dcol * SEQ + jb + kk) >> 1];
      }
      acc = __builtin_amdgcn_wmma_f32_16x16x32_bf16(
          false, __builtin_bit_cast(v16bf, au),
          false, __builtin_bit_cast(v16bf, bu),
          (short)0, acc, false, false);
    }
#pragma unroll
    for (int r = 0; r < 8; ++r) {
      int n = row0 + wm * 16 + half * 8 + r;
      obf[(size_t)(b * SEQ + n) * DIM + h * HD + wn * 16 + nl] = f2bfbits(acc[r]);
    }
  }
}

// ---------------------------------------------------------------------------
// Kernel 3: out = O @ out_w^T + out_b  (f32 output; both operands bf16)
// ---------------------------------------------------------------------------
__global__ __launch_bounds__(256) void out_proj_kernel(
    const unsigned short* __restrict__ obf, const unsigned short* __restrict__ Wbf,
    const float* __restrict__ bout, float* __restrict__ out)
{
  const int tid  = threadIdx.x;
  const int wave = tid >> 5, lane = tid & 31;
  const int wm = wave >> 2, wn = wave & 3;
  const int half = lane >> 4, nl = lane & 15;
  const int row0 = blockIdx.y * 32;
  const int col0 = blockIdx.x * 64;
  const int rA = row0 + wm * 16 + nl;
  const int cB = col0 + wn * 16 + nl;
  const unsigned* arow = (const unsigned*)(obf + (size_t)rA * DIM);
  const unsigned* wrow = (const unsigned*)(Wbf + (size_t)cB * DIM);

  v8f acc = {};
  for (int kb = 0; kb < DIM; kb += 32) {
    v8u au, bu;
#pragma unroll
    for (int p = 0; p < 8; ++p) {
      int k = kb + kOfPair(p, half);
      au[p] = arow[k >> 1];
      bu[p] = wrow[k >> 1];
    }
    acc = __builtin_amdgcn_wmma_f32_16x16x32_bf16(
        false, __builtin_bit_cast(v16bf, au),
        false, __builtin_bit_cast(v16bf, bu),
        (short)0, acc, false, false);
  }
  const float bc = bout[cB];
#pragma unroll
  for (int r = 0; r < 8; ++r) {
    int m = row0 + wm * 16 + half * 8 + r;
    out[(size_t)m * DIM + cB] = acc[r] + bc;
  }
}

extern "C" void kernel_launch(void* const* d_in, const int* in_sizes, int n_in,
                              void* d_out, int out_size, void* d_ws, size_t ws_size,
                              hipStream_t stream) {
  const float* x            = (const float*)d_in[0];
  const unsigned char* mask = (const unsigned char*)d_in[1];   // numpy bool
  const float* Wqkv = (const float*)d_in[2];
  const float* bqkv = (const float*)d_in[3];
  const float* qn_g = (const float*)d_in[4];
  const float* qn_b = (const float*)d_in[5];
  const float* kn_g = (const float*)d_in[6];
  const float* kn_b = (const float*)d_in[7];
  const float* Wout = (const float*)d_in[8];
  const float* bout = (const float*)d_in[9];
  float* out = (float*)d_out;

  char* ws = (char*)d_ws;
  const size_t MB = 1024 * 1024;
  unsigned short* qws    = (unsigned short*)(ws);              //  8 MB
  unsigned short* kws    = (unsigned short*)(ws +  8 * MB);    //  8 MB
  unsigned short* vT     = (unsigned short*)(ws + 16 * MB);    //  8 MB
  unsigned short* obf    = (unsigned short*)(ws + 24 * MB);    //  8 MB
  unsigned short* xbf    = (unsigned short*)(ws + 32 * MB);    //  8 MB
  unsigned short* wqkvbf = (unsigned short*)(ws + 40 * MB);    // 1.5 MB
  unsigned short* woutbf = (unsigned short*)(ws + 42 * MB);    // 0.5 MB

  dim3 blk(256);
  // One-time operand conversions (bandwidth-bound, ~21 MB total)
  cvt_bf16_kernel<<<1024, blk, 0, stream>>>(x,    (unsigned*)xbf,    BATCH * SEQ * DIM / 2);
  cvt_bf16_kernel<<<256,  blk, 0, stream>>>(Wqkv, (unsigned*)wqkvbf, QKVD * DIM / 2);
  cvt_bf16_kernel<<<128,  blk, 0, stream>>>(Wout, (unsigned*)woutbf, DIM * DIM / 2);

  qkv_ln_rope_kernel<<<dim3(QKVD / 64, (BATCH * SEQ) / 32), blk, 0, stream>>>(
      xbf, wqkvbf, bqkv, qn_g, qn_b, kn_g, kn_b, qws, kws, vT);

  size_t smem = (size_t)32 * SEQ * sizeof(float);   // 256 KB (CDNA5: 320 KB/WG)
  attn_kernel<<<dim3(SEQ / 32, BATCH * NH), blk, smem, stream>>>(
      qws, kws, vT, mask, obf);

  out_proj_kernel<<<dim3(DIM / 64, (BATCH * SEQ) / 32), blk, 0, stream>>>(
      obf, woutbf, bout, out);
}
